// GCN_PyG_67259187855731
// MI455X (gfx1250) — matrix operand order repaired and checked by reading, not compiled, not run on hardware
//
#include <hip/hip_runtime.h>
#include <hip/hip_bf16.h>

typedef __attribute__((ext_vector_type(16))) __bf16 v16bf;
typedef __attribute__((ext_vector_type(2)))  __bf16 v2bf;
typedef __attribute__((ext_vector_type(8)))  float  v8f;
typedef __attribute__((ext_vector_type(4)))  float  v4f;

#define IN_DIM 256
#define HID    32
#define OUT    12

// ---- bf16 conversion helpers -------------------------------------------------
__device__ __forceinline__ unsigned short bf_bits(float f) {   // RNE (one-time LDS staging)
    unsigned u = __builtin_bit_cast(unsigned, f);
    u += 0x7fffu + ((u >> 16) & 1u);
    return (unsigned short)(u >> 16);
}

// hot-path pair conversion: round-half-up + v_perm_b32 pack (3 VALU per pair)
__device__ __forceinline__ v2bf cvt2(float lo, float hi) {
#if __has_builtin(__builtin_amdgcn_cvt_pk_bf16_f32)
    return __builtin_amdgcn_cvt_pk_bf16_f32(lo, hi);
#elif __has_builtin(__builtin_amdgcn_perm)
    unsigned ulo = __builtin_bit_cast(unsigned, lo) + 0x8000u;
    unsigned uhi = __builtin_bit_cast(unsigned, hi) + 0x8000u;
    // D = { uhi[31:16], ulo[31:16] }  (bytes: 2,3 of ulo ; 6,7 of {uhi,ulo})
    unsigned packed = __builtin_amdgcn_perm(uhi, ulo, 0x07060302u);
    return __builtin_bit_cast(v2bf, packed);
#else
    v2bf r;
    r.x = __builtin_bit_cast(__bf16, bf_bits(lo));
    r.y = __builtin_bit_cast(__bf16, bf_bits(hi));
    return r;
#endif
}

union AFrag { v16bf v; v2bf p[8]; };

__device__ __forceinline__ v16bf make_afrag(v4f q0, v4f q1, v4f q2, v4f q3) {
    AFrag a;
    a.p[0] = cvt2(q0.x, q0.y); a.p[1] = cvt2(q0.z, q0.w);
    a.p[2] = cvt2(q1.x, q1.y); a.p[3] = cvt2(q1.z, q1.w);
    a.p[4] = cvt2(q2.x, q2.y); a.p[5] = cvt2(q2.z, q2.w);
    a.p[6] = cvt2(q3.x, q3.y); a.p[7] = cvt2(q3.z, q3.w);
    return a.v;
}

// ---- small elementwise / edge kernels ---------------------------------------
__global__ void fill_f32(float* __restrict__ p, int n, float v) {
    int i = blockIdx.x * blockDim.x + threadIdx.x;
    if (i < n) p[i] = v;
}

__global__ void deg_accum(const int* __restrict__ dst, float* __restrict__ deg, int E) {
    int e = blockIdx.x * blockDim.x + threadIdx.x;
    if (e < E) atomicAdd(&deg[dst[e]], 1.0f);
}

__global__ void deg_rsqrt(float* __restrict__ dinv, int n) {
    int i = blockIdx.x * blockDim.x + threadIdx.x;
    if (i < n) dinv[i] = rsqrtf(dinv[i]);   // deg >= 1 (self loop)
}

__global__ void edge_norm(const int* __restrict__ src, const int* __restrict__ dst,
                          const float* __restrict__ dinv, float* __restrict__ nrm, int E) {
    int e = blockIdx.x * blockDim.x + threadIdx.x;
    if (e < E) nrm[e] = dinv[src[e]] * dinv[dst[e]];
}

// ---------------- GEMM1: h1[N,32] = x[N,256] @ W1[256,32]  (bf16 WMMA, f32 acc) ----
// W1 staged in LDS as bf16, transposed to [col][k] so each B fragment is one
// contiguous 32B span (2x ds_load_b128). x rows streamed with NT hints (read-once;
// keep the 192MB L2 for the graph working set). One wave -> 16x32 tile, 8 K-steps.
__global__ __launch_bounds__(256) void gemm1_wmma(const float* __restrict__ x,
                                                  const float* __restrict__ W1,
                                                  float* __restrict__ h1, int ntiles) {
    __shared__ alignas(32) unsigned short wlds[HID][IN_DIM];   // 16KB bf16 bits
    for (int idx = threadIdx.x; idx < IN_DIM * HID; idx += 256) {
        int k = idx >> 5, col = idx & 31;                      // W1 row-major [k][col]
        wlds[col][k] = bf_bits(W1[idx]);
    }
    __syncthreads();

    const int lane = threadIdx.x & 31;
    const int wave = threadIdx.x >> 5;
    const int tile = blockIdx.x * 8 + wave;                    // wave-uniform guard
    if (tile >= ntiles) return;

    const int mrow = lane & 15;
    const int kh   = lane >> 4;
    const float* xr = x + (size_t)(tile * 16 + mrow) * IN_DIM;

    v8f c0 = {}; v8f c1 = {};
    #pragma unroll
    for (int kb = 0; kb < IN_DIM / 32; ++kb) {
        // A 16x32 bf16: lanes 0-15 K=0..15 ; lanes 16-31 K=16..31
        const int k0 = kb * 32 + kh * 16;
        v4f q0 = __builtin_nontemporal_load((const v4f*)(xr + k0));
        v4f q1 = __builtin_nontemporal_load((const v4f*)(xr + k0 + 4));
        v4f q2 = __builtin_nontemporal_load((const v4f*)(xr + k0 + 8));
        v4f q3 = __builtin_nontemporal_load((const v4f*)(xr + k0 + 12));
        const v16bf a = make_afrag(q0, q1, q2, q3);

        // B 32x16: lanes 0-15 -> K rows kb*32+0..15, lanes 16-31 -> +16..31
        const int krB = kb * 32 + kh * 16;
        const v16bf b0 = *reinterpret_cast<const v16bf*>(&wlds[mrow][krB]);
        const v16bf b1 = *reinterpret_cast<const v16bf*>(&wlds[mrow + 16][krB]);

        c0 = __builtin_amdgcn_wmma_f32_16x16x32_bf16(false, a, false, b0, (short)0, c0, false, false);
        c1 = __builtin_amdgcn_wmma_f32_16x16x32_bf16(false, a, false, b1, (short)0, c1, false, false);
    }
    // C layout: VGPR r -> M=r (lanes 0-15) / M=r+8 (lanes 16-31); N = lane&15
    const int rbase = tile * 16 + kh * 8;
    #pragma unroll
    for (int r = 0; r < 8; ++r) {
        float* o = h1 + (size_t)(rbase + r) * HID;
        o[mrow]      = c0[r];
        o[mrow + 16] = c1[r];
    }
}

// scatter layer 1: 8 threads per edge, float4 gather (L2-resident) + 4 atomics
__global__ void scatter1(const int* __restrict__ src, const int* __restrict__ dst,
                         const float* __restrict__ nrm, const float* __restrict__ h1,
                         float* __restrict__ a1, int E) {
    int gid = blockIdx.x * blockDim.x + threadIdx.x;
    int e = gid >> 3;
    if (e >= E) return;
    int c = (gid & 7) * 4;
    int s = src[e], d = dst[e];
    float w = nrm[e];
    const float4 v = *(const float4*)(h1 + (size_t)s * HID + c);
    float* o = a1 + (size_t)d * HID + c;
    atomicAdd(o + 0, w * v.x);
    atomicAdd(o + 1, w * v.y);
    atomicAdd(o + 2, w * v.z);
    atomicAdd(o + 3, w * v.w);
}

// z = relu(agg + dinv^2 * h_self + b1), in place into a1
__global__ void node1(float* __restrict__ a1, const float* __restrict__ h1,
                      const float* __restrict__ dinv, const float* __restrict__ b1, int n) {
    int gid = blockIdx.x * blockDim.x + threadIdx.x;   // n = N*32
    if (gid >= n) return;
    int i = gid >> 5, j = gid & 31;
    float di = dinv[i];
    float v = a1[gid] + di * di * h1[gid] + b1[j];
    a1[gid] = fmaxf(v, 0.0f);
}

// ---------------- GEMM2: h2[N,12] = z[N,32] @ W2[32,12] — one WMMA per tile -------
__global__ __launch_bounds__(256) void gemm2_wmma(const float* __restrict__ z,
                                                  const float* __restrict__ W2,
                                                  float* __restrict__ h2, int ntiles) {
    __shared__ alignas(32) unsigned short wlds[16][HID];       // 1KB, cols padded 12->16
    for (int idx = threadIdx.x; idx < 16 * HID; idx += 256) {
        int col = idx >> 5, k = idx & 31;
        wlds[col][k] = (col < OUT) ? bf_bits(W2[(size_t)k * OUT + col]) : (unsigned short)0;
    }
    __syncthreads();

    const int lane = threadIdx.x & 31;
    const int wave = threadIdx.x >> 5;
    const int tile = blockIdx.x * 8 + wave;
    if (tile >= ntiles) return;

    const int mrow = lane & 15;
    const int kh   = lane >> 4;
    const float* zr = z + (size_t)(tile * 16 + mrow) * HID + kh * 16;

    v4f q0 = *(const v4f*)(zr);
    v4f q1 = *(const v4f*)(zr + 4);
    v4f q2 = *(const v4f*)(zr + 8);
    v4f q3 = *(const v4f*)(zr + 12);
    const v16bf a = make_afrag(q0, q1, q2, q3);

    const v16bf b = *reinterpret_cast<const v16bf*>(&wlds[mrow][kh * 16]);

    v8f c = {};
    c = __builtin_amdgcn_wmma_f32_16x16x32_bf16(false, a, false, b, (short)0, c, false, false);

    if (mrow < OUT) {
        const int rbase = tile * 16 + kh * 8;
        #pragma unroll
        for (int r = 0; r < 8; ++r)
            h2[(size_t)(rbase + r) * OUT + mrow] = c[r];
    }
}

// scatter layer 2: one thread per (edge, feature)
__global__ void scatter2(const int* __restrict__ src, const int* __restrict__ dst,
                         const float* __restrict__ nrm, const float* __restrict__ h2,
                         float* __restrict__ a2, int E) {
    int gid = blockIdx.x * blockDim.x + threadIdx.x;
    int e = gid / OUT;
    if (e >= E) return;
    int j = gid - e * OUT;
    atomicAdd(&a2[(size_t)dst[e] * OUT + j], nrm[e] * h2[(size_t)src[e] * OUT + j]);
}

__global__ void node2(const float* __restrict__ a2, const float* __restrict__ h2,
                      const float* __restrict__ dinv, const float* __restrict__ b2,
                      float* __restrict__ out, int n) {
    int gid = blockIdx.x * blockDim.x + threadIdx.x;   // n = N*12
    if (gid >= n) return;
    int i = gid / OUT, j = gid - i * OUT;
    float di = dinv[i];
    out[gid] = a2[gid] + di * di * h2[gid] + b2[j];
}

extern "C" void kernel_launch(void* const* d_in, const int* in_sizes, int n_in,
                              void* d_out, int out_size, void* d_ws, size_t ws_size,
                              hipStream_t stream) {
    const float* x   = (const float*)d_in[0];
    const int*   ei  = (const int*)  d_in[1];
    const float* W1  = (const float*)d_in[2];
    const float* b1  = (const float*)d_in[3];
    const float* W2  = (const float*)d_in[4];
    const float* b2  = (const float*)d_in[5];
    float* out = (float*)d_out;

    const int N = in_sizes[0] / IN_DIM;     // 100000
    const int E = in_sizes[1] / 2;          // 3200000
    const int* src = ei;
    const int* dst = ei + E;

    // workspace layout (floats): dinv | h1 | a1 | h2 | a2 | norm  (~48 MB)
    float* dinv = (float*)d_ws;
    float* h1   = dinv + N;
    float* a1   = h1 + (size_t)N * HID;
    float* h2   = a1 + (size_t)N * HID;
    float* a2   = h2 + (size_t)N * OUT;
    float* nrm  = a2 + (size_t)N * OUT;

    const int T = 256;
    auto gs = [](long long n, int t) { return (unsigned)((n + t - 1) / t); };

    // init: deg=1 (self loop), zero accumulators
    fill_f32<<<gs(N, T), T, 0, stream>>>(dinv, N, 1.0f);
    fill_f32<<<gs((long long)N * HID, T), T, 0, stream>>>(a1, N * HID, 0.0f);
    fill_f32<<<gs((long long)N * OUT, T), T, 0, stream>>>(a2, N * OUT, 0.0f);

    // degrees -> dinv -> per-edge norm
    deg_accum<<<gs(E, T), T, 0, stream>>>(dst, dinv, E);
    deg_rsqrt<<<gs(N, T), T, 0, stream>>>(dinv, N);
    edge_norm<<<gs(E, T), T, 0, stream>>>(src, dst, dinv, nrm, E);

    // layer 1
    const int ntiles = (N + 15) / 16;       // 6250 full tiles
    gemm1_wmma<<<gs(ntiles, 8), T, 0, stream>>>(x, W1, h1, ntiles);
    scatter1<<<gs((long long)E * 8, T), T, 0, stream>>>(src, dst, nrm, h1, a1, E);
    node1<<<gs((long long)N * HID, T), T, 0, stream>>>(a1, h1, dinv, b1, N * HID);

    // layer 2
    gemm2_wmma<<<gs(ntiles, 8), T, 0, stream>>>(a1, W2, h2, ntiles);
    scatter2<<<gs((long long)E * OUT, T), T, 0, stream>>>(src, dst, nrm, h2, a2, E);
    node2<<<gs((long long)N * OUT, T), T, 0, stream>>>(a2, h2, dinv, b2, out, N * OUT);
}
